// CrossModalAdaptiveFusion_66984309949198
// MI455X (gfx1250) — matrix operand order repaired and checked by reading, not compiled, not run on hardware
//
#include <hip/hip_runtime.h>
#include <hip/hip_bf16.h>

// ---------------------------------------------------------------------------
// CrossModalAdaptiveFusion for MI455X (gfx1250, wave32, WMMA + TDM)
// ---------------------------------------------------------------------------

#define C_DIM   768
#define S_TOT   32768           // 32*32*32 spatial
#define GROUPS  12
#define GSIZE   64              // channels per group
#define KTAPS   27

typedef __bf16 bf16_t;
typedef __attribute__((ext_vector_type(16))) __bf16 v16bf;
typedef __attribute__((ext_vector_type(8)))  __bf16 v8bf;
typedef __attribute__((ext_vector_type(8)))  float  v8f;

typedef unsigned int u32;
typedef unsigned short u16;
typedef u32 __attribute__((ext_vector_type(4))) u32x4;
typedef int __attribute__((ext_vector_type(4))) i32x4;
typedef int __attribute__((ext_vector_type(8))) i32x8;

#ifndef HAVE_TDM
#if defined(__has_builtin)
#if __has_builtin(__builtin_amdgcn_tensor_load_to_lds) && __has_builtin(__builtin_amdgcn_s_wait_tensorcnt)
#define HAVE_TDM 1
#endif
#endif
#endif
#ifndef HAVE_TDM
#define HAVE_TDM 0
#endif

// ---------------------------------------------------------------------------
// 1) per-channel spatial mean (vc) -> combined[0:768]
// ---------------------------------------------------------------------------
__global__ __launch_bounds__(256) void colmean_kernel(const float* __restrict__ vf,
                                                      float* __restrict__ outc) {
  __shared__ float red[256];
  const int c = blockIdx.x, tid = threadIdx.x;
  const float4* p = (const float4*)(vf + (size_t)c * S_TOT);
  float s = 0.f;
  for (int i = tid; i < S_TOT / 4; i += 256) {
    float4 v = p[i];
    s += v.x + v.y + v.z + v.w;
  }
  red[tid] = s;
  __syncthreads();
  for (int st = 128; st > 0; st >>= 1) {
    if (tid < st) red[tid] += red[tid + st];
    __syncthreads();
  }
  if (tid == 0) outc[c] = red[0] * (1.f / (float)S_TOT);
}

// ---------------------------------------------------------------------------
// 2) generic GEMV: y[row] = act(sum_c W[row,c]*x[c] + b[row])
//    one wave (32 lanes) per output row.  act: 0 none, 1 relu, 2 sigmoid
// ---------------------------------------------------------------------------
__global__ __launch_bounds__(256) void gemv_kernel(const float* __restrict__ W,
                                                   const float* __restrict__ x,
                                                   const float* __restrict__ b,
                                                   float* __restrict__ y,
                                                   int in_dim, int act) {
  const int g = blockIdx.x * blockDim.x + threadIdx.x;
  const int row = g >> 5;
  const int lane = g & 31;
  const float4* w4 = (const float4*)(W + (size_t)row * in_dim);
  const float4* x4 = (const float4*)x;
  const int n4 = in_dim >> 2;
  float s = 0.f;
  for (int i = lane; i < n4; i += 32) {
    float4 a = w4[i], c = x4[i];
    s += a.x * c.x + a.y * c.y + a.z * c.z + a.w * c.w;
  }
  for (int off = 16; off > 0; off >>= 1) s += __shfl_xor(s, off);
  if (lane == 0) {
    s += b[row];
    if (act == 1) s = fmaxf(s, 0.f);
    else if (act == 2) s = 1.f / (1.f + __expf(-s));
    y[row] = s;
  }
}

// ---------------------------------------------------------------------------
// 3) dynamic depthwise 3x3x3 conv; channel modulation folded into the taps
//    (sum k*(m*x) == sum (k*m)*x).  The 2-3 x-row slab (8-12KB, contiguous in
//    memory) is DMA'd into LDS by the Tensor Data Mover; edge rows zeroed.
// ---------------------------------------------------------------------------
__global__ __launch_bounds__(256) void dwconv_kernel(const float* __restrict__ vf,
                                                     const float* __restrict__ kp,
                                                     const float* __restrict__ modv,
                                                     float* __restrict__ out) {
  __shared__ __align__(16) float slab[3][32][32];       // [x-1,x,x+1][y][z]
  const int c = blockIdx.x;
  const int x = blockIdx.y;
  const int tid = threadIdx.x;
  float* slabf = &slab[0][0][0];

  const float m = modv[c];
  float kk[KTAPS];
#pragma unroll
  for (int t = 0; t < KTAPS; ++t) kk[t] = kp[c * KTAPS + t] * m;

  const int xi0 = (x == 0) ? 0 : (x - 1);
  const int xi1 = (x == 31) ? 31 : (x + 1);
  const int nrows = xi1 - xi0 + 1;              // 2 at edges, 3 interior

  // zero the slab row the DMA will not cover
  if (x == 0)
    for (int i = tid; i < 1024; i += 256) slabf[i] = 0.f;
  if (x == 31)
    for (int i = tid; i < 1024; i += 256) slabf[2 * 1024 + i] = 0.f;

#if HAVE_TDM
  if (tid < 32) {                                // wave 0 issues the TDM op
    const unsigned long long ga =
        (unsigned long long)(uintptr_t)(vf + ((size_t)c * 32 + xi0) * 1024);
    const u32 lds_byte =
        (u32)(uintptr_t)slabf + ((x == 0) ? 4096u : 0u);
    u32x4 g0;
    g0.x = 1u;                                   // count=1 (valid descriptor)
    g0.y = lds_byte;                             // lds_addr (bytes)
    g0.z = (u32)ga;                              // global_addr[31:0]
    g0.w = (u32)(ga >> 32) | 0x80000000u;        // global_addr[56:32] | type=2
    i32x8 g1;
    g1[0] = 0x00020000;                          // data_size = 4 bytes
    g1[1] = (int)(1024u << 16);                  // tensor_dim0 = 1024
    g1[2] = (int)((u32)nrows << 16);             // tensor_dim1 = nrows
    g1[3] = (int)(1024u << 16);                  // tile_dim0 = 1024
    g1[4] = nrows;                               // tile_dim1 = nrows
    g1[5] = 1024;                                // tensor_dim0_stride = 1024
    g1[6] = 0;
    g1[7] = 0;
    i32x4 z4 = {0, 0, 0, 0};
#if __clang_major__ >= 23
    i32x8 z8 = {0, 0, 0, 0, 0, 0, 0, 0};
    __builtin_amdgcn_tensor_load_to_lds(g0, g1, z4, z4, z8, 0);
#else
    __builtin_amdgcn_tensor_load_to_lds(g0, g1, z4, z4, 0);
#endif
    __builtin_amdgcn_s_wait_tensorcnt(0);
  }
#else
  for (int i = tid; i < nrows * 1024; i += 256)
    slabf[((x == 0) ? 1024 : 0) + i] = vf[((size_t)c * 32 + xi0) * 1024 + i];
#endif
  __syncthreads();

  for (int i = tid; i < 1024; i += 256) {
    const int y = i >> 5, z = i & 31;
    float s = 0.f;
#pragma unroll
    for (int a = 0; a < 3; ++a)
#pragma unroll
      for (int b = 0; b < 3; ++b) {
        const int yy = y + b - 1;
        if (yy < 0 || yy >= 32) continue;
#pragma unroll
        for (int d = 0; d < 3; ++d) {
          const int zz = z + d - 1;
          if (zz < 0 || zz >= 32) continue;
          s += kk[a * 9 + b * 3 + d] * slab[a][yy][zz];
        }
      }
    out[((size_t)c * 32 + x) * 1024 + i] = s;
  }
}

// ---------------------------------------------------------------------------
// 4) per-channel sum / sum-of-squares for GroupNorm
// ---------------------------------------------------------------------------
__global__ __launch_bounds__(256) void chanstat_kernel(const float* __restrict__ x,
                                                       float* __restrict__ cs,
                                                       float* __restrict__ cq) {
  __shared__ float rs[256], rq[256];
  const int c = blockIdx.x, tid = threadIdx.x;
  const float4* p = (const float4*)(x + (size_t)c * S_TOT);
  float s = 0.f, q = 0.f;
  for (int i = tid; i < S_TOT / 4; i += 256) {
    float4 v = p[i];
    s += v.x + v.y + v.z + v.w;
    q += v.x * v.x + v.y * v.y + v.z * v.z + v.w * v.w;
  }
  rs[tid] = s; rq[tid] = q;
  __syncthreads();
  for (int st = 128; st > 0; st >>= 1) {
    if (tid < st) { rs[tid] += rs[tid + st]; rq[tid] += rq[tid + st]; }
    __syncthreads();
  }
  if (tid == 0) { cs[c] = rs[0]; cq[c] = rq[0]; }
}

// ---------------------------------------------------------------------------
// 5) fold GroupNorm into per-channel affine: xn = out*scale[c] + shift[c]
// ---------------------------------------------------------------------------
__global__ void gn_affine_kernel(const float* __restrict__ cs, const float* __restrict__ cq,
                                 const float* __restrict__ gw, const float* __restrict__ gb,
                                 float* __restrict__ scale, float* __restrict__ shift) {
  const int c = threadIdx.x;                 // 768 threads
  const int g0 = (c / GSIZE) * GSIZE;
  float s = 0.f, q = 0.f;
  for (int j = 0; j < GSIZE; ++j) { s += cs[g0 + j]; q += cq[g0 + j]; }
  const float invN = 1.f / ((float)GSIZE * (float)S_TOT);
  const float mean = s * invN;
  const float var = q * invN - mean * mean;
  const float rstd = rsqrtf(var + 1e-5f);
  const float sc = rstd * gw[c];
  scale[c] = sc;
  shift[c] = gb[c] - mean * sc;
}

// ---------------------------------------------------------------------------
// 6) apply affine + convert activations / weights to bf16
// ---------------------------------------------------------------------------
__global__ __launch_bounds__(256) void cvt_x_kernel(const float* __restrict__ xin,
                                                    const float* __restrict__ scale,
                                                    const float* __restrict__ shift,
                                                    bf16_t* __restrict__ xb) {
  const size_t i4 = (size_t)blockIdx.x * blockDim.x + threadIdx.x;
  const int c = (int)(i4 >> 13);             // 8192 float4 per channel
  const float sc = scale[c], sh = shift[c];
  const float4 v = ((const float4*)xin)[i4];
  const size_t o = i4 * 4;
  xb[o + 0] = (bf16_t)(v.x * sc + sh);
  xb[o + 1] = (bf16_t)(v.y * sc + sh);
  xb[o + 2] = (bf16_t)(v.z * sc + sh);
  xb[o + 3] = (bf16_t)(v.w * sc + sh);
}

__global__ __launch_bounds__(256) void cvt_w_kernel(const float* __restrict__ w,
                                                    bf16_t* __restrict__ wb) {
  const int i = blockIdx.x * blockDim.x + threadIdx.x;   // 589824 elements
  wb[i] = (bf16_t)w[i];
}

// ---------------------------------------------------------------------------
// 7) final 1x1x1 projection as bf16 WMMA GEMM (f32 accumulate):
//    Y[768 x 32768] = Wb[768 x 768] * Xn[768 x 32768] + bias
//    Block: 256 thr / 8 waves -> 64(M) x 256(N); wave = 16x128 (8 acc tiles).
//    Double-buffered LDS; transposed staging uses packed b64 DS stores.
// ---------------------------------------------------------------------------
#define LDSK 40                  // bf16 per LDS row (32 + pad), 80B keeps 16B align
#define BN   256                 // block N
#define WN   128                 // wave N

__global__ __launch_bounds__(256) void wmma_gemm_kernel(const bf16_t* __restrict__ Wb,
                                                        const bf16_t* __restrict__ Xb,
                                                        const float* __restrict__ bias,
                                                        float* __restrict__ Y) {
  __shared__ __align__(16) bf16_t ldsx[2][BN * LDSK];   // X tile transposed [n][k], 2x20KB

  const int tid = threadIdx.x;
  const int wave = tid >> 5;
  const int lane = tid & 31;
  const int half = lane >> 4;        // WMMA lane-half
  const int l16 = lane & 15;
  const int wave_m = wave >> 1;      // 0..3
  const int wave_n = wave & 1;       // 0..1
  const int n0 = blockIdx.x * BN;
  const int m0 = blockIdx.y * 64;
  const int mrow = m0 + wave_m * 16 + l16;        // A-fragment row (M = lane%16)
  const bf16_t* wbase = Wb + (size_t)mrow * C_DIM;

  // staging: each thread owns a 4(k) x 8(n) sub-block
  const int kq = (tid & 7) * 4;      // k base: 0..28
  const int nch = (tid >> 3) * 8;    // n base: 0..248

  v8f acc[8];
#pragma unroll
  for (int t = 0; t < 8; ++t) acc[t] = (v8f){0.f, 0.f, 0.f, 0.f, 0.f, 0.f, 0.f, 0.f};

  uint4 ld[4];

  auto load_tile = [&](int kbase) {
#pragma unroll
    for (int j = 0; j < 4; ++j)
      ld[j] = *(const uint4*)(Xb + (size_t)(kbase + kq + j) * S_TOT + n0 + nch);
  };
  auto store_tile = [&](int buf) {
    const u16* u0 = (const u16*)&ld[0];
    const u16* u1 = (const u16*)&ld[1];
    const u16* u2 = (const u16*)&ld[2];
    const u16* u3 = (const u16*)&ld[3];
#pragma unroll
    for (int i = 0; i < 8; ++i) {            // 8 packed b64 stores (4 bf16 each)
      uint2 pk;
      pk.x = (u32)u0[i] | ((u32)u1[i] << 16);
      pk.y = (u32)u2[i] | ((u32)u3[i] << 16);
      *(uint2*)&ldsx[buf][(size_t)(nch + i) * LDSK + kq] = pk;
    }
  };

  load_tile(0);
  store_tile(0);
  __syncthreads();

  for (int it = 0; it < C_DIM / 32; ++it) {
    const int k0 = it * 32;
    const int buf = it & 1;
    const bool more = (it + 1) < C_DIM / 32;
    if (more) load_tile(k0 + 32);            // overlap next-tile global loads

    // A fragment: W[mrow, k0 + half*8 + {0..7, 16..23}]
    union { v16bf v; v8bf h[2]; } a;
    const bf16_t* wrow = wbase + k0 + half * 8;
    a.h[0] = *(const v8bf*)(wrow);
    a.h[1] = *(const v8bf*)(wrow + 16);
    if (more) __builtin_prefetch(wrow + 32, 0, 0);   // global_prefetch_b8

    // 8 N-subtiles reuse the A fragment
#pragma unroll
    for (int t = 0; t < 8; ++t) {
      union { v16bf v; v8bf h[2]; } b;
      const bf16_t* brow =
          &ldsx[buf][(size_t)(wave_n * WN + t * 16 + l16) * LDSK + half * 8];
      b.h[0] = *(const v8bf*)(brow);         // K = half*8 + 0..7
      b.h[1] = *(const v8bf*)(brow + 16);    // K = 16 + half*8 + 0..7
      acc[t] = __builtin_amdgcn_wmma_f32_16x16x32_bf16(
          false, a.v, false, b.v, (short)0, acc[t], false, false);
    }

    if (more) store_tile(buf ^ 1);           // fill the other buffer
    __syncthreads();                         // one barrier per K step
  }

  // store D: acc[t][r] holds M = m0 + wave_m*16 + 8*half + r, N = lane%16
#pragma unroll
  for (int t = 0; t < 8; ++t) {
    const int col = n0 + wave_n * WN + t * 16 + l16;
#pragma unroll
    for (int r = 0; r < 8; ++r) {
      const int orow = m0 + wave_m * 16 + 8 * half + r;
      Y[(size_t)orow * S_TOT + col] = acc[t][r] + bias[orow];
    }
  }
}

// ---------------------------------------------------------------------------
// host side
// ---------------------------------------------------------------------------
extern "C" void kernel_launch(void* const* d_in, const int* in_sizes, int n_in,
                              void* d_out, int out_size, void* d_ws, size_t ws_size,
                              hipStream_t stream) {
  const float* vf   = (const float*)d_in[0];
  const float* text = (const float*)d_in[1];
  const float* tpw  = (const float*)d_in[2];
  const float* tpb  = (const float*)d_in[3];
  const float* ipw  = (const float*)d_in[4];
  const float* ipb  = (const float*)d_in[5];
  const float* opw  = (const float*)d_in[6];
  const float* opb  = (const float*)d_in[7];
  const float* kw1  = (const float*)d_in[8];
  const float* kb1  = (const float*)d_in[9];
  const float* kw2  = (const float*)d_in[10];
  const float* kb2  = (const float*)d_in[11];
  const float* mw   = (const float*)d_in[12];
  const float* mb   = (const float*)d_in[13];
  const float* gw   = (const float*)d_in[14];
  const float* gb   = (const float*)d_in[15];
  const float* cw   = (const float*)d_in[16];
  const float* cb   = (const float*)d_in[17];
  float* Y = (float*)d_out;

  char* ws = (char*)d_ws;
  float*  conv_out = (float*)(ws);                         // 768*32768*4  = 100663296 B
  bf16_t* xnb      = (bf16_t*)(ws + 100663296);            // 768*32768*2  =  50331648 B
  bf16_t* Wb       = (bf16_t*)(ws + 150994944);            // 768*768*2    =   1179648 B
  float*  sm       = (float*)(ws + 152174592);             // small vectors
  float* combined = sm;            // 1536  ([vc, attn_context])
  float* tp       = sm + 1536;     // 768
  float* vvec     = sm + 2304;     // 768
  float* h1       = sm + 3072;     // 3072
  float* kp       = sm + 6144;     // 20736
  float* modv     = sm + 26880;    // 768
  float* csum     = sm + 27648;    // 768
  float* csq      = sm + 28416;    // 768
  float* scale    = sm + 29184;    // 768
  float* shift    = sm + 29952;    // 768

  // vc -> combined[0:768]
  colmean_kernel<<<768, 256, 0, stream>>>(vf, combined);
  // tp = text_proj(text)
  gemv_kernel<<<96, 256, 0, stream>>>(tpw, text, tpb, tp, C_DIM, 0);
  // v = in_proj[2C:](tp)   (softmax over single key == 1 -> attn == v)
  gemv_kernel<<<96, 256, 0, stream>>>(ipw + 2 * C_DIM * C_DIM, tp, ipb + 2 * C_DIM,
                                      vvec, C_DIM, 0);
  // attn_context -> combined[768:1536]
  gemv_kernel<<<96, 256, 0, stream>>>(opw, vvec, opb, combined + C_DIM, C_DIM, 0);
  // h1 = relu(kn_w1 @ combined + b)
  gemv_kernel<<<384, 256, 0, stream>>>(kw1, combined, kb1, h1, 2 * C_DIM, 1);
  // kp = kn_w2 @ h1 + b          (20736 rows, bandwidth bound: 255 MB)
  gemv_kernel<<<2592, 256, 0, stream>>>(kw2, h1, kb2, kp, 4 * C_DIM, 0);
  // mod = sigmoid(mod_w @ combined + b)
  gemv_kernel<<<96, 256, 0, stream>>>(mw, combined, mb, modv, 2 * C_DIM, 2);
  // dynamic depthwise conv (TDM slab staging, modulation folded into taps)
  dwconv_kernel<<<dim3(768, 32), 256, 0, stream>>>(vf, kp, modv, conv_out);
  // GroupNorm folded into per-channel affine
  chanstat_kernel<<<768, 256, 0, stream>>>(conv_out, csum, csq);
  gn_affine_kernel<<<1, 768, 0, stream>>>(csum, csq, gw, gb, scale, shift);
  // bf16 conversion for the WMMA GEMM
  cvt_x_kernel<<<24576, 256, 0, stream>>>(conv_out, scale, shift, xnb);
  cvt_w_kernel<<<2304, 256, 0, stream>>>(cw, Wb);
  // Y = conv_w @ xn + conv_b   (38.7 GFLOP -> v_wmma_f32_16x16x32_bf16)
  wmma_gemm_kernel<<<dim3(S_TOT / BN, C_DIM / 64), 256, 0, stream>>>(Wb, xnb, cb, Y);
}
